// GTNN_outer_15625091022926
// MI455X (gfx1250) — matrix operand order repaired and checked by reading, not compiled, not run on hardware
//
#include <hip/hip_runtime.h>
#include <stdint.h>

// ---- problem constants (match reference) ----
#define DIMF 128     // node feature dim
#define GSF  64      // SAGE out dim
#define AFF  32      // ir_score dim
#define TGS  128     // 2*GS
#define KBIG 16384   // (2*GS)^2
#define HID  200     // MLP hidden
#define NBC  2       // classes

typedef float  v8f   __attribute__((ext_vector_type(8)));
typedef __bf16 v16bf __attribute__((ext_vector_type(16)));

// ---------------------------------------------------------------------------
// K1: per-graph node counts
__global__ void k_count(const int* __restrict__ batch_ids, int* __restrict__ counts, int N) {
    int i = blockIdx.x * blockDim.x + threadIdx.x;
    if (i < N) atomicAdd(&counts[batch_ids[i]], 1);
}

// ---------------------------------------------------------------------------
// K2: exclusive scan over counts (single 1024-thread block), then sib + slot maps
__global__ void k_scan_sib(const int* __restrict__ counts, const int* __restrict__ set_indices,
                           int* __restrict__ bases, int* __restrict__ sib,
                           int* __restrict__ slotA, int* __restrict__ slotB, int G) {
    __shared__ int part[1024];
    int t = threadIdx.x;
    int per = (G + 1023) / 1024;
    int base = t * per;
    int s = 0;
    for (int j = 0; j < per; ++j) { int g = base + j; if (g < G) s += counts[g]; }
    part[t] = s;
    __syncthreads();
    for (int off = 1; off < 1024; off <<= 1) {
        int v = part[t];
        int add = (t >= off) ? part[t - off] : 0;
        __syncthreads();
        part[t] = v + add;
        __syncthreads();
    }
    int run = (t == 0) ? 0 : part[t - 1];
    for (int j = 0; j < per; ++j) {
        int g = base + j;
        if (g < G) { bases[g] = run; run += counts[g]; }
    }
    __syncthreads();
    for (int g = t; g < G; g += 1024) {
        int b  = bases[g];
        int n0 = b + set_indices[g * 2 + 0];
        int n1 = b + set_indices[g * 2 + 1];
        sib[g * 2 + 0] = n0;
        sib[g * 2 + 1] = n1;
        slotA[n0] = g * 2 + 0;   // unique writer per (g, s) -> deterministic
        slotB[n1] = g * 2 + 1;   // handles sib0==sib1 (separate arrays)
    }
}

// ---------------------------------------------------------------------------
// K3: edge scatter-add, filtered to selected destinations. One wave per edge.
__global__ void k_edge(const int* __restrict__ ei, const float* __restrict__ x,
                       const int* __restrict__ slotA, const int* __restrict__ slotB,
                       float* __restrict__ aggSlot, int* __restrict__ degSlot, int E) {
    int wid  = (blockIdx.x * blockDim.x + threadIdx.x) >> 5;
    int lane = threadIdx.x & 31;
    if (wid >= E) return;
    int src = ei[wid];
    int dst = ei[E + wid];
    int sa = slotA[dst];
    int sb = slotB[dst];
    if (sa < 0 && sb < 0) return;                       // wave-uniform
    const float4 xv = *(const float4*)(x + (size_t)src * DIMF + lane * 4);
    if (sa >= 0) {
        float* p = aggSlot + (size_t)sa * DIMF + lane * 4;
        atomicAdd(p + 0, xv.x); atomicAdd(p + 1, xv.y);
        atomicAdd(p + 2, xv.z); atomicAdd(p + 3, xv.w);
        if (lane == 0) atomicAdd(&degSlot[sa], 1);
    }
    if (sb >= 0) {
        float* p = aggSlot + (size_t)sb * DIMF + lane * 4;
        atomicAdd(p + 0, xv.x); atomicAdd(p + 1, xv.y);
        atomicAdd(p + 2, xv.z); atomicAdd(p + 3, xv.w);
        if (lane == 0) atomicAdd(&degSlot[sb], 1);
    }
}

// ---------------------------------------------------------------------------
// K4: h row for each selected slot: relu(mean@W_l + b_l + x@W_r) -> pair[g, s*64..]
__global__ void k_pair(const float* __restrict__ x, const float* __restrict__ aggSlot,
                       const int* __restrict__ degSlot, const int* __restrict__ sib,
                       const float* __restrict__ W_l, const float* __restrict__ b_l,
                       const float* __restrict__ W_r, float* __restrict__ pairO) {
    __shared__ float mS[DIMF];
    __shared__ float xS[DIMF];
    int slot = blockIdx.x;
    int t = threadIdx.x;                 // 64 threads
    int node = sib[slot];
    float dg = (float)degSlot[slot];
    dg = dg > 1.f ? dg : 1.f;
    for (int j = t; j < DIMF; j += 64) {
        mS[j] = aggSlot[(size_t)slot * DIMF + j] / dg;
        xS[j] = x[(size_t)node * DIMF + j];
    }
    __syncthreads();
    float acc = b_l[t];
    #pragma unroll 8
    for (int k = 0; k < DIMF; ++k)
        acc += mS[k] * W_l[k * GSF + t] + xS[k] * W_r[k * GSF + t];
    int g = slot >> 1, s = slot & 1;
    pairO[(size_t)g * TGS + s * GSF + t] = acc > 0.f ? acc : 0.f;
}

// ---------------------------------------------------------------------------
// K5: S = relu(ir_score @ W_ir + b_ir)
__global__ void k_S(const float* __restrict__ ir, const float* __restrict__ W_ir,
                    const float* __restrict__ b_ir, float* __restrict__ Sout) {
    __shared__ float irS[AFF];
    int g = blockIdx.x;
    int o = threadIdx.x;                 // 128 threads
    if (o < AFF) irS[o] = ir[(size_t)g * AFF + o];
    __syncthreads();
    float acc = b_ir[o];
    #pragma unroll
    for (int k = 0; k < AFF; ++k) acc += irS[k] * W_ir[k * TGS + o];
    Sout[(size_t)g * TGS + o] = acc > 0.f ? acc : 0.f;
}

// ---------------------------------------------------------------------------
// K6: W1T[n,k] = bf16(W1[k,n]), n padded to 256 with zeros
__global__ void k_w1t(const float* __restrict__ W1, __bf16* __restrict__ W1T) {
    int idx = blockIdx.x * blockDim.x + threadIdx.x;   // 256*16384 total
    int n = idx >> 14;
    int k = idx & (KBIG - 1);
    __bf16 v = (__bf16)0.f;
    if (n < HID) v = (__bf16)W1[(size_t)k * HID + n];
    W1T[(size_t)n * KBIG + k] = v;
}

// ---------------------------------------------------------------------------
// K7: z = relu((pair (x) S) @ W1 + b1), fused rank-1 A-tile build + bf16 WMMA.
// One wave per (16-graph x 32-col) strip: ONE A-frag feeds TWO WMMAs, halving
// the VALU A-build cost per WMMA so it fits the XDL co-execution shadow.
// grid = (G/16, 7), block = 32.
__global__ void k_gemm(const float* __restrict__ pairM, const float* __restrict__ SM,
                       const __bf16* __restrict__ W1T, const float* __restrict__ b1,
                       float* __restrict__ z) {
    const int gbase = blockIdx.x * 16;
    const int nbase = blockIdx.y * 32;   // two adjacent 16-col tiles
    const int lane  = threadIdx.x;
    const int row   = lane & 15;
    const int half  = lane >> 4;         // 0 or 1
    const int khalf = half * 8;

    const float*  Srow  = SM    + (size_t)(gbase + row) * TGS;
    const float*  Prow  = pairM + (size_t)(gbase + row) * TGS;
    const __bf16* Brow0 = W1T + (size_t)(nbase + row) * KBIG;       // col n0 = nbase + (lane&15)
    const __bf16* Brow1 = W1T + (size_t)(nbase + 16 + row) * KBIG;  // col n1 = n0 + 16

    // Hoist all S values this lane needs (4 j-chunks x 16 elems), in A-frag order:
    //   elems 0..15 <-> j = jc*32 + 16*half + e
    float sv[4][16];
    #pragma unroll
    for (int jc = 0; jc < 4; ++jc) {
        const float4 s0 = *(const float4*)(Srow + jc * 32 + half * 16 + 0);
        const float4 s1 = *(const float4*)(Srow + jc * 32 + half * 16 + 4);
        const float4 s2 = *(const float4*)(Srow + jc * 32 + half * 16 + 8);
        const float4 s3 = *(const float4*)(Srow + jc * 32 + half * 16 + 12);
        sv[jc][0] = s0.x;  sv[jc][1] = s0.y;  sv[jc][2]  = s0.z;  sv[jc][3]  = s0.w;
        sv[jc][4] = s1.x;  sv[jc][5] = s1.y;  sv[jc][6]  = s1.z;  sv[jc][7]  = s1.w;
        sv[jc][8] = s2.x;  sv[jc][9] = s2.y;  sv[jc][10] = s2.z;  sv[jc][11] = s2.w;
        sv[jc][12] = s3.x; sv[jc][13] = s3.y; sv[jc][14] = s3.z;  sv[jc][15] = s3.w;
    }

    v8f acc0 = {};
    v8f acc1 = {};
    for (int i = 0; i < TGS; ++i) {
        const float a = Prow[i];
        const int kr = i * TGS;                        // k row base in W1T
        __builtin_prefetch(Brow0 + kr + TGS, 0, 0);    // global_prefetch_b8: next B rows
        __builtin_prefetch(Brow1 + kr + TGS, 0, 0);
        #pragma unroll
        for (int jc = 0; jc < 4; ++jc) {
            const int kc = kr + jc * 32;
            // B frags: elems 0..15 <-> K = 16*half + 0..15 (contiguous 32B each)
            v16bf bfrag0 = *(const v16bf*)(Brow0 + kc + half * 16);
            v16bf bfrag1 = *(const v16bf*)(Brow1 + kc + half * 16);
            // A frag: rank-1 product built once, reused by both WMMAs
            v16bf afrag;
            #pragma unroll
            for (int t = 0; t < 16; ++t) afrag[t] = (__bf16)(a * sv[jc][t]);
            acc0 = __builtin_amdgcn_wmma_f32_16x16x32_bf16(
                false, afrag, false, bfrag0, (short)0, acc0, false, false);
            acc1 = __builtin_amdgcn_wmma_f32_16x16x32_bf16(
                false, afrag, false, bfrag1, (short)0, acc1, false, false);
        }
    }

    // Epilogue: +b1, ReLU, store. C/D layout: VGPR r <-> M = half*8 + r, N = lane&15.
    const int n0 = nbase + row;
    const int n1 = nbase + 16 + row;
    if (n0 < HID) {
        const float bb = b1[n0];
        #pragma unroll
        for (int r = 0; r < 8; ++r) {
            const int m = gbase + half * 8 + r;
            float v = acc0[r] + bb;
            z[(size_t)m * HID + n0] = v > 0.f ? v : 0.f;
        }
    }
    if (n1 < HID) {
        const float bb = b1[n1];
        #pragma unroll
        for (int r = 0; r < 8; ++r) {
            const int m = gbase + half * 8 + r;
            float v = acc1[r] + bb;
            z[(size_t)m * HID + n1] = v > 0.f ? v : 0.f;
        }
    }
}

// ---------------------------------------------------------------------------
// K8: logits = z @ W2 + b2. One wave per graph.
__global__ void k_decode(const float* __restrict__ z, const float* __restrict__ W2,
                         const float* __restrict__ b2, float* __restrict__ out, int G) {
    int wid  = (blockIdx.x * blockDim.x + threadIdx.x) >> 5;
    int lane = threadIdx.x & 31;
    if (wid >= G) return;
    const float* zr = z + (size_t)wid * HID;
    float a0 = 0.f, a1 = 0.f;
    for (int k = lane; k < HID; k += 32) {
        float v = zr[k];
        a0 += v * W2[k * NBC + 0];
        a1 += v * W2[k * NBC + 1];
    }
    #pragma unroll
    for (int off = 16; off > 0; off >>= 1) {
        a0 += __shfl_down(a0, off, 32);
        a1 += __shfl_down(a1, off, 32);
    }
    if (lane == 0) {
        out[(size_t)wid * NBC + 0] = a0 + b2[0];
        out[(size_t)wid * NBC + 1] = a1 + b2[1];
    }
}

// ---------------------------------------------------------------------------
static inline size_t alignup(size_t x) { return (x + 255) & ~(size_t)255; }

extern "C" void kernel_launch(void* const* d_in, const int* in_sizes, int n_in,
                              void* d_out, int out_size, void* d_ws, size_t ws_size,
                              hipStream_t stream) {
    const float* x        = (const float*)d_in[0];
    const int*   ei       = (const int*)  d_in[1];
    const int*   set_idx  = (const int*)  d_in[2];
    const int*   batch_id = (const int*)  d_in[3];
    const float* ir       = (const float*)d_in[4];
    const float* W_l      = (const float*)d_in[5];
    const float* b_l      = (const float*)d_in[6];
    const float* W_r      = (const float*)d_in[7];
    const float* W_ir     = (const float*)d_in[8];
    const float* b_ir     = (const float*)d_in[9];
    const float* W1       = (const float*)d_in[10];
    const float* b1       = (const float*)d_in[11];
    const float* W2       = (const float*)d_in[12];
    const float* b2       = (const float*)d_in[13];
    float* out = (float*)d_out;

    const int E = in_sizes[1] / 2;
    const int G = in_sizes[2] / 2;
    const int N = in_sizes[3];
    const int NSLOT = 2 * G;

    // workspace carve-up
    char* w = (char*)d_ws;
    size_t off = 0;
    int*   counts  = (int*)(w + off);   off = alignup(off + (size_t)G * 4);
    int*   bases   = (int*)(w + off);   off = alignup(off + (size_t)G * 4);
    int*   sib     = (int*)(w + off);   off = alignup(off + (size_t)NSLOT * 4);
    int*   slotA   = (int*)(w + off);   off = alignup(off + (size_t)N * 4);
    int*   slotB   = (int*)(w + off);   off = alignup(off + (size_t)N * 4);
    int*   degSlot = (int*)(w + off);   off = alignup(off + (size_t)NSLOT * 4);
    float* aggSlot = (float*)(w + off); off = alignup(off + (size_t)NSLOT * DIMF * 4);
    float* pairM   = (float*)(w + off); off = alignup(off + (size_t)G * TGS * 4);
    float* SM      = (float*)(w + off); off = alignup(off + (size_t)G * TGS * 4);
    __bf16* W1T    = (__bf16*)(w + off);off = alignup(off + (size_t)256 * KBIG * 2);
    float* z       = (float*)(w + off); off = alignup(off + (size_t)G * HID * 4);
    (void)ws_size; (void)n_in; (void)out_size;

    hipMemsetAsync(counts,  0,    (size_t)G * 4,              stream);
    hipMemsetAsync(slotA,   0xFF, (size_t)N * 4,              stream);
    hipMemsetAsync(slotB,   0xFF, (size_t)N * 4,              stream);
    hipMemsetAsync(degSlot, 0,    (size_t)NSLOT * 4,          stream);
    hipMemsetAsync(aggSlot, 0,    (size_t)NSLOT * DIMF * 4,   stream);

    k_count<<<(N + 255) / 256, 256, 0, stream>>>(batch_id, counts, N);
    k_scan_sib<<<1, 1024, 0, stream>>>(counts, set_idx, bases, sib, slotA, slotB, G);
    k_edge<<<(E + 7) / 8, 256, 0, stream>>>(ei, x, slotA, slotB, aggSlot, degSlot, E);
    k_pair<<<NSLOT, 64, 0, stream>>>(x, aggSlot, degSlot, sib, W_l, b_l, W_r, pairM);
    k_S<<<G, TGS, 0, stream>>>(ir, W_ir, b_ir, SM);
    k_w1t<<<(256 * KBIG) / 256, 256, 0, stream>>>(W1, W1T);

    dim3 gg(G / 16, 7);   // 256 cols padded: 7 strips of 32 cols (rows 0..223 of W1T)
    k_gemm<<<gg, 32, 0, stream>>>(pairM, SM, W1T, b1, z);

    k_decode<<<(G + 7) / 8, 256, 0, stream>>>(z, W2, b2, out, G);
}
